// decoder_main_encoder_VQ_FC_74088185856416
// MI455X (gfx1250) — matrix operand order repaired and checked by reading, not compiled, not run on hardware
//
#include <hip/hip_runtime.h>
#include <hip/hip_bf16.h>

typedef __attribute__((ext_vector_type(16))) _Float16 v16h;
typedef __attribute__((ext_vector_type(8)))  float    v8f;

#define D_      512
#define NTOK_   256
#define HEADS_  16
#define DH_     32
#define DEPTH_  12
#define KCODE_  2048
#define NB_     16
#define QLD_    1536

// ---------------------------------------------------------------------------
// CDNA5 async global->LDS copy (ASYNCcnt path, CDNA5 ISA).
// Per lane: LDS[vdst] = MEM[vaddr] (16 bytes). Low 32 bits of a flat LDS
// pointer are the LDS byte offset (ISA §10.2 aperture rules).
// ---------------------------------------------------------------------------
__device__ inline void async_copy_b128(const _Float16* gptr, _Float16* sptr) {
  unsigned lds = (unsigned)(size_t)sptr;
  asm volatile("global_load_async_to_lds_b128 %0, %1, off"
               :: "v"(lds), "v"(gptr) : "memory");
}
__device__ inline void async_wait_all() {
  asm volatile("s_wait_asynccnt 0" ::: "memory");
}

// ---------------------------------------------------------------------------
// WMMA fragment loaders (wave32; A/B use this file's contiguous k order, C the ISA layout)
// A 16x32 f16: lane l holds row (l&15); element e -> K = e + 16*(l>>4)
// B 32x16 f16: lane l holds col (l&15); element e -> K = e + 16*(l>>4)
// C 16x16 f32: vgpr r, lane l -> M = r + 8*(l>>4), N = (l&15)
// ---------------------------------------------------------------------------
__device__ inline v16h frag_a_lds(const _Float16* s, int ld, int row0, int lane) {
  v16h a;
  const _Float16* p = s + (size_t)(row0 + (lane & 15)) * ld + ((lane >> 4) << 4);
#pragma unroll
  for (int e = 0; e < 16; ++e) a[e] = p[e];
  return a;
}

__device__ inline v16h frag_b_lds(const _Float16* s, int ld, int col0, int lane) {
  v16h b;
  const _Float16* p = s + (size_t)((lane >> 4) << 4) * ld + col0 + (lane & 15);
#pragma unroll
  for (int e = 0; e < 16; ++e) b[e] = p[(size_t)e * ld];
  return b;
}

// ---------------------------------------------------------------------------
// f16-in GEMM via WMMA, 64x128 block tile, K step 32, 4 waves x 8 WMMA/step.
// M multiple of 64, K multiple of 32 and N multiple of 128 (true for every call).
// C[M,N] = A[M,K] * op(B) (+bias[N]) (+res[M,N]); BT: B stored [N,K].
// OUTH: write f16 output (Ch), else f32 (C).
// ---------------------------------------------------------------------------
template <bool BIAS, bool RES, bool BT, bool OUTH>
__global__ __launch_bounds__(128) void gemm_h(
    const _Float16* __restrict__ A, int lda, const _Float16* __restrict__ B, int ldb,
    const float* __restrict__ bias, const float* __restrict__ res,
    float* __restrict__ C, _Float16* __restrict__ Ch, int ldc, int M, int N, int K) {
  __shared__ _Float16 As[64 * 40];
  __shared__ _Float16 Bs[32 * 136];
  const int tid = threadIdx.x, lane = tid & 31, wave = tid >> 5;
  const int wm = wave >> 1, wn = wave & 1;
  const int m0 = blockIdx.x * 64, n0 = blockIdx.y * 128;
  v8f acc[2][4] = {};
  for (int k0 = 0; k0 < K; k0 += 32) {
    // A tile 64x32 halves: async b128 global->LDS (ASYNCcnt)
#pragma unroll
    for (int ch = tid; ch < 256; ch += 128) {
      int r = ch >> 2, c = (ch & 3) * 8;
      async_copy_b128(&A[(size_t)(m0 + r) * lda + k0 + c], &As[r * 40 + c]);
    }
    // B tile 32x128 halves
    if (!BT) {
#pragma unroll
      for (int ch = tid; ch < 512; ch += 128) {
        int r = ch >> 4, c = (ch & 15) * 8;
        async_copy_b128(&B[(size_t)(k0 + r) * ldb + n0 + c], &Bs[r * 136 + c]);
      }
    } else {
      for (int ch = tid; ch < 512; ch += 128) {
        int r = ch >> 4, c = (ch & 15) * 8;
#pragma unroll
        for (int j = 0; j < 8; ++j)
          Bs[r * 136 + c + j] = B[(size_t)(n0 + c + j) * ldb + k0 + r];
      }
    }
    async_wait_all();
    __syncthreads();
    if (k0 + 32 < K)
      __builtin_prefetch(&A[(size_t)(m0 + (tid >> 1)) * lda + k0 + 32], 0, 3);
    v16h a0 = frag_a_lds(As, 40, wm * 32, lane);
    v16h a1 = frag_a_lds(As, 40, wm * 32 + 16, lane);
#pragma unroll
    for (int j = 0; j < 4; ++j) {
      v16h b = frag_b_lds(Bs, 136, wn * 64 + j * 16, lane);
      acc[0][j] = __builtin_amdgcn_wmma_f32_16x16x32_f16(false, a0, false, b,
                                                         (short)0, acc[0][j],
                                                         false, false);
      acc[1][j] = __builtin_amdgcn_wmma_f32_16x16x32_f16(false, a1, false, b,
                                                         (short)0, acc[1][j],
                                                         false, false);
    }
    __syncthreads();
  }
#pragma unroll
  for (int i = 0; i < 2; ++i)
#pragma unroll
    for (int j = 0; j < 4; ++j)
#pragma unroll
      for (int r = 0; r < 8; ++r) {
        int row = m0 + wm * 32 + i * 16 + r + ((lane >> 4) << 3);
        int col = n0 + wn * 64 + j * 16 + (lane & 15);
        float v = acc[i][j][r];
        if (BIAS) v += bias[col];
        if (RES) v += res[(size_t)row * ldc + col];
        if (OUTH)
          Ch[(size_t)row * ldc + col] = (_Float16)v;
        else
          C[(size_t)row * ldc + col] = v;
      }
}

// ---------------------------------------------------------------------------
// Attention (f16 qkv in, f16 out): one WG (16 waves) per (batch, head).
// Wave w owns query rows w*16..w*16+15. DH=32 == one WMMA K-step.
// ---------------------------------------------------------------------------
__global__ __launch_bounds__(512) void attn_wmma(const _Float16* __restrict__ qkv,
                                                 _Float16* __restrict__ obuf) {
  constexpr int VLD = 40, PLD = 40;
  __shared__ _Float16 Vs[256 * VLD];
  __shared__ _Float16 Pst[16 * 16 * PLD];
  const int tid = threadIdx.x, lane = tid & 31, wave = tid >> 5;
  const int bi = blockIdx.x >> 4, h = blockIdx.x & 15;
  const _Float16* base = qkv + (size_t)bi * NTOK_ * QLD_ + h * DH_;
  // stage V[key][d]: async b128 copies (256*32 halves -> 1024 chunks)
#pragma unroll
  for (int ch = tid; ch < 1024; ch += 512) {
    int key = ch >> 2, c = (ch & 3) * 8;
    async_copy_b128(&base[(size_t)key * QLD_ + 2 * D_ + c], &Vs[key * VLD + c]);
  }
  async_wait_all();
  __syncthreads();
  const int q0 = wave * 16;
  v16h aq;
  {
    const _Float16* qp = base + (size_t)(q0 + (lane & 15)) * QLD_;
    int kb = (lane >> 4) << 4;
#pragma unroll
    for (int e = 0; e < 16; ++e) aq[e] = qp[e + kb];
  }
  // S = Q K^T : B fragment per lane = 16 contiguous halves of one K row
  v8f s[16];
  const _Float16* kbase = base + D_;
#pragma unroll
  for (int t = 0; t < 16; ++t) {
    v16h bk;
    const _Float16* kp =
        kbase + (size_t)(t * 16 + (lane & 15)) * QLD_ + ((lane >> 4) << 4);
#pragma unroll
    for (int e = 0; e < 16; ++e) bk[e] = kp[e];
    v8f z = {};
    s[t] = __builtin_amdgcn_wmma_f32_16x16x32_f16(false, aq, false, bk, (short)0,
                                                  z, false, false);
  }
  // softmax (rows 0..7 on lanes 0..15, rows 8..15 on lanes 16..31)
  const float scale = 0.17677669529663687f;  // 1/sqrt(32)
#pragma unroll
  for (int r = 0; r < 8; ++r) {
    float mx = -3.4e38f;
#pragma unroll
    for (int t = 0; t < 16; ++t) {
      s[t][r] *= scale;
      mx = fmaxf(mx, s[t][r]);
    }
#pragma unroll
    for (int m = 1; m < 16; m <<= 1) mx = fmaxf(mx, __shfl_xor(mx, m, 32));
    float sum = 0.f;
#pragma unroll
    for (int t = 0; t < 16; ++t) {
      float e = __expf(s[t][r] - mx);
      s[t][r] = e;
      sum += e;
    }
#pragma unroll
    for (int m = 1; m < 16; m <<= 1) sum += __shfl_xor(sum, m, 32);
    float inv = 1.f / sum;
#pragma unroll
    for (int t = 0; t < 16; ++t) s[t][r] *= inv;
  }
  // O = P V: C-layout -> A-layout transpose through per-wave LDS staging
  _Float16* pst = Pst + wave * 16 * PLD;
  v8f o0 = {}, o1 = {};
#pragma unroll
  for (int t2 = 0; t2 < 8; ++t2) {
#pragma unroll
    for (int r = 0; r < 8; ++r) {
      int row = r + ((lane >> 4) << 3);
      pst[row * PLD + (lane & 15)]      = (_Float16)s[2 * t2][r];
      pst[row * PLD + 16 + (lane & 15)] = (_Float16)s[2 * t2 + 1][r];
    }
    asm volatile("s_wait_dscnt 0" ::: "memory");  // wave-local LDS RAW fence
    v16h ap  = frag_a_lds(pst, PLD, 0, lane);
    v16h bv0 = frag_b_lds(Vs + t2 * 32 * VLD, VLD, 0, lane);
    v16h bv1 = frag_b_lds(Vs + t2 * 32 * VLD, VLD, 16, lane);
    o0 = __builtin_amdgcn_wmma_f32_16x16x32_f16(false, ap, false, bv0, (short)0,
                                                o0, false, false);
    o1 = __builtin_amdgcn_wmma_f32_16x16x32_f16(false, ap, false, bv1, (short)0,
                                                o1, false, false);
  }
  _Float16* op = obuf + ((size_t)bi * NTOK_ + q0) * D_ + h * DH_;
#pragma unroll
  for (int r = 0; r < 8; ++r) {
    int row = r + ((lane >> 4) << 3);
    op[(size_t)row * D_ + (lane & 15)]      = (_Float16)o0[r];
    op[(size_t)row * D_ + 16 + (lane & 15)] = (_Float16)o1[r];
  }
}

// ---------------------------------------------------------------------------
// Implicit-GEMM 3x3 conv (pad 1): M=Cout, N=N*H*W pixels (mult of 128),
// K=Cin per tap. Weights pre-repacked f16 [tap][Co][Ci] (contiguous Ci).
// Output f32 (feeds batch-norm statistics).
// ---------------------------------------------------------------------------
__global__ __launch_bounds__(128) void conv3_h(const _Float16* __restrict__ in,
                                               const _Float16* __restrict__ wt9,
                                               float* __restrict__ out, int Nb,
                                               int Ci, int Co, int H, int W) {
  __shared__ _Float16 As[64 * 40];
  __shared__ _Float16 Bs[32 * 136];
  const int tid = threadIdx.x, lane = tid & 31, wave = tid >> 5;
  const int wm = wave >> 1, wn = wave & 1;
  const int co0 = blockIdx.x * 64, p0 = blockIdx.y * 128;
  const int HW = H * W;
  v8f acc[2][4] = {};
  for (int tap = 0; tap < 9; ++tap) {
    const int kh = tap / 3, kw = tap - kh * 3;
    const _Float16* wtap = wt9 + (size_t)tap * Co * Ci;
    for (int k0 = 0; k0 < Ci; k0 += 32) {
      // weight tile: async b128 global->LDS
#pragma unroll
      for (int ch = tid; ch < 256; ch += 128) {
        int r = ch >> 2, c = (ch & 3) * 8;
        async_copy_b128(&wtap[(size_t)(co0 + r) * Ci + k0 + c], &As[r * 40 + c]);
      }
      // pixel tile: halo-guarded scalar gather
      for (int i = tid; i < 32 * 128; i += 128) {
        int r = i >> 7, c = i & 127;
        int p = p0 + c;
        int n = p / HW, hw = p - n * HW;
        int hh = hw / W, ww = hw - hh * W;
        int ih = hh + kh - 1, iw = ww + kw - 1;
        _Float16 v = (_Float16)0.f;
        if (ih >= 0 && ih < H && iw >= 0 && iw < W)
          v = in[(((size_t)n * Ci + k0 + r) * H + ih) * W + iw];
        Bs[r * 136 + c] = v;
      }
      async_wait_all();
      __syncthreads();
      v16h a0 = frag_a_lds(As, 40, wm * 32, lane);
      v16h a1 = frag_a_lds(As, 40, wm * 32 + 16, lane);
#pragma unroll
      for (int j = 0; j < 4; ++j) {
        v16h b = frag_b_lds(Bs, 136, wn * 64 + j * 16, lane);
        acc[0][j] = __builtin_amdgcn_wmma_f32_16x16x32_f16(false, a0, false, b,
                                                           (short)0, acc[0][j],
                                                           false, false);
        acc[1][j] = __builtin_amdgcn_wmma_f32_16x16x32_f16(false, a1, false, b,
                                                           (short)0, acc[1][j],
                                                           false, false);
      }
      __syncthreads();
    }
  }
  const int NHW = Nb * HW;
#pragma unroll
  for (int i = 0; i < 2; ++i)
#pragma unroll
    for (int j = 0; j < 4; ++j)
#pragma unroll
      for (int r = 0; r < 8; ++r) {
        int co = co0 + wm * 32 + i * 16 + r + ((lane >> 4) << 3);
        int p = p0 + wn * 64 + j * 16 + (lane & 15);
        if (co < Co && p < NHW) {
          int n = p / HW, hw = p - n * HW;
          out[((size_t)n * Co + co) * HW + hw] = acc[i][j][r];
        }
      }
}

// ---------------------------------------------------------------------------
// Elementwise / reduction kernels
// ---------------------------------------------------------------------------
__global__ void cvt_f16(const float* __restrict__ s, _Float16* __restrict__ d,
                        long n) {
  long i = (long)blockIdx.x * blockDim.x + threadIdx.x;
  if (i < n) d[i] = (_Float16)s[i];
}

// repack [Co,Ci,3,3] f32 -> [tap][Co][Ci] f16
__global__ void repack_w9(const float* __restrict__ src, _Float16* __restrict__ dst,
                          int Co, int Ci) {
  long total = (long)Co * Ci * 9;
  long id = (long)blockIdx.x * blockDim.x + threadIdx.x;
  if (id >= total) return;
  int tap = id % 9;
  long t = id / 9;
  int ci = t % Ci;
  int co = (int)(t / Ci);
  dst[((size_t)tap * Co + co) * Ci + ci] = (_Float16)src[id];
}

__global__ void tok_init(const float* __restrict__ x, const float* __restrict__ pos,
                         float* __restrict__ tok) {
  long id = (long)blockIdx.x * blockDim.x + threadIdx.x;
  long total = (long)NB_ * NTOK_ * D_;
  if (id >= total) return;
  int d = id % D_;
  long t = id / D_;
  int s = t % NTOK_;
  int n = (int)(t / NTOK_);
  tok[id] = x[((size_t)n * D_ + d) * NTOK_ + s] + pos[(size_t)s * D_ + d];
}

__global__ __launch_bounds__(256) void ln_kernel(const float* __restrict__ x,
                                                 const float* __restrict__ g,
                                                 const float* __restrict__ b,
                                                 _Float16* __restrict__ y) {
  int row = blockIdx.x, tid = threadIdx.x;
  const float* xp = x + (size_t)row * D_;
  float v0 = xp[tid], v1 = xp[tid + 256];
  __shared__ float ss[256], sq[256];
  ss[tid] = v0 + v1;
  sq[tid] = v0 * v0 + v1 * v1;
  __syncthreads();
  for (int st = 128; st > 0; st >>= 1) {
    if (tid < st) { ss[tid] += ss[tid + st]; sq[tid] += sq[tid + st]; }
    __syncthreads();
  }
  float m = ss[0] * (1.f / D_);
  float var = sq[0] * (1.f / D_) - m * m;
  float inv = rsqrtf(var + 1e-5f);
  _Float16* yp = y + (size_t)row * D_;
  yp[tid] = (_Float16)((v0 - m) * inv * g[tid] + b[tid]);
  yp[tid + 256] = (_Float16)((v1 - m) * inv * g[tid + 256] + b[tid + 256]);
}

__global__ __launch_bounds__(128) void code_norm(const float* __restrict__ cb,
                                                 float* __restrict__ cn) {
  int c = blockIdx.x, tid = threadIdx.x;
  const float* p = cb + (size_t)c * D_;
  float s = 0.f;
  for (int i = tid; i < D_; i += 128) s += p[i] * p[i];
  __shared__ float ss[128];
  ss[tid] = s;
  __syncthreads();
  for (int st = 64; st > 0; st >>= 1) {
    if (tid < st) ss[tid] += ss[tid + st];
    __syncthreads();
  }
  if (tid == 0) cn[c] = ss[0];
}

__global__ __launch_bounds__(256) void argmin_kernel(const float* __restrict__ dots,
                                                     const float* __restrict__ cnorm,
                                                     int* __restrict__ idx) {
  int row = blockIdx.x, tid = threadIdx.x;
  const float* dp = dots + (size_t)row * KCODE_;
  float best = 3.4e38f;
  int bi = 0;
  for (int c = tid; c < KCODE_; c += 256) {
    float v = cnorm[c] - 2.f * dp[c];
    if (v < best) { best = v; bi = c; }
  }
  __shared__ float bv[256];
  __shared__ int bx[256];
  bv[tid] = best;
  bx[tid] = bi;
  __syncthreads();
  for (int st = 128; st > 0; st >>= 1) {
    if (tid < st) {
      if (bv[tid + st] < bv[tid] ||
          (bv[tid + st] == bv[tid] && bx[tid + st] < bx[tid])) {
        bv[tid] = bv[tid + st];
        bx[tid] = bx[tid + st];
      }
    }
    __syncthreads();
  }
  if (tid == 0) idx[row] = bx[0];
}

// gather selected codes (f16) into NCHW token grid for the decoder
__global__ void gather_q(const _Float16* __restrict__ cbh,
                         const int* __restrict__ idx, _Float16* __restrict__ y0) {
  long id = (long)blockIdx.x * blockDim.x + threadIdx.x;
  long total = (long)NB_ * NTOK_ * D_;
  if (id >= total) return;
  int d = id % D_;
  int token = (int)(id / D_);
  int n = token >> 8, s = token & 255;
  y0[(((size_t)n * D_ + d) << 8) + s] = cbh[(size_t)idx[token] * D_ + d];
}

// ConvTranspose2d k=2,s=2 (torch weight [Cin,Cout,2,2], f32); f16 in/out; writes
// into concat buffer with channel stride Ctot.
__global__ void convt2_h(const _Float16* __restrict__ x, const float* __restrict__ wt,
                         const float* __restrict__ bias, _Float16* __restrict__ out,
                         int Nb, int Ci, int Co, int H, int W, int Ctot) {
  int H2 = 2 * H, W2 = 2 * W;
  long total = (long)Nb * Co * H2 * W2;
  long id = (long)blockIdx.x * blockDim.x + threadIdx.x;
  if (id >= total) return;
  int w2 = id % W2;
  long t = id / W2;
  int h2 = t % H2;
  t /= H2;
  int co = t % Co;
  int n = (int)(t / Co);
  int i = h2 & 1, j = w2 & 1, hh = h2 >> 1, ww = w2 >> 1;
  float acc = bias[co];
  const _Float16* xp = x + (size_t)n * Ci * H * W + hh * W + ww;
  for (int ci = 0; ci < Ci; ++ci)
    acc += (float)xp[(size_t)ci * H * W] * wt[(((size_t)ci * Co + co) * 2 + i) * 2 + j];
  out[(((size_t)n * Ctot + co) * H2 + h2) * W2 + w2] = (_Float16)acc;
}

__global__ void copy_skip_h(const float* __restrict__ f, _Float16* __restrict__ u,
                            int Nb, int Cs, int Coff, int Ctot, int HW) {
  long total = (long)Nb * Cs * HW;
  long id = (long)blockIdx.x * blockDim.x + threadIdx.x;
  if (id >= total) return;
  int hw = id % HW;
  long t = id / HW;
  int c = t % Cs;
  int n = (int)(t / Cs);
  u[((size_t)n * Ctot + Coff + c) * HW + hw] = (_Float16)f[id];
}

__global__ __launch_bounds__(256) void bn_stats(const float* __restrict__ x,
                                                float* __restrict__ mean,
                                                float* __restrict__ inv, int Nb,
                                                int C, int HW) {
  int c = blockIdx.x, tid = threadIdx.x;
  float s = 0.f, sq = 0.f;
  for (int n = 0; n < Nb; ++n) {
    const float* p = x + ((size_t)n * C + c) * HW;
    for (int i = tid; i < HW; i += 256) {
      float v = p[i];
      s += v;
      sq += v * v;
    }
  }
  __shared__ float ss[256], sqq[256];
  ss[tid] = s;
  sqq[tid] = sq;
  __syncthreads();
  for (int st = 128; st > 0; st >>= 1) {
    if (tid < st) { ss[tid] += ss[tid + st]; sqq[tid] += sqq[tid + st]; }
    __syncthreads();
  }
  if (tid == 0) {
    float cnt = (float)Nb * HW;
    float m = ss[0] / cnt;
    float v = sqq[0] / cnt - m * m;
    mean[c] = m;
    inv[c] = rsqrtf(v + 1e-5f);
  }
}

__global__ void bn_apply_relu_h(const float* __restrict__ x,
                                const float* __restrict__ mean,
                                const float* __restrict__ inv,
                                const float* __restrict__ g,
                                const float* __restrict__ b,
                                _Float16* __restrict__ y, int C, int HW,
                                long total) {
  long i = (long)blockIdx.x * blockDim.x + threadIdx.x;
  if (i >= total) return;
  int c = (int)((i / HW) % C);
  float v = (x[i] - mean[c]) * inv[c] * g[c] + b[c];
  y[i] = (_Float16)(v > 0.f ? v : 0.f);
}

__global__ void final_conv_sigmoid(const _Float16* __restrict__ x,
                                   const float* __restrict__ wt,
                                   const float* __restrict__ bias,
                                   float* __restrict__ out, int Nb, int C, int HW) {
  long total = (long)Nb * HW;
  long id = (long)blockIdx.x * blockDim.x + threadIdx.x;
  if (id >= total) return;
  int hw = id % HW;
  int n = (int)(id / HW);
  const _Float16* xp = x + (size_t)n * C * HW + hw;
  float acc = bias[0];
  for (int c = 0; c < C; ++c) acc += (float)xp[(size_t)c * HW] * wt[c];
  out[id] = 1.f / (1.f + __expf(-acc));
}

// ---------------------------------------------------------------------------
// Host orchestration
// ---------------------------------------------------------------------------
static inline long cdl(long a, long b) { return (a + b - 1) / b; }

extern "C" void kernel_launch(void* const* d_in, const int* in_sizes, int n_in,
                              void* d_out, int out_size, void* d_ws, size_t ws_size,
                              hipStream_t stream) {
  const float* x     = (const float*)d_in[0];
  const float* feat0 = (const float*)d_in[1];
  const float* feat1 = (const float*)d_in[2];
  const float* feat2 = (const float*)d_in[3];
  const float* pos   = (const float*)d_in[4];
  const float* cb    = (const float*)d_in[5];
  const float* ln_g  = (const float*)d_in[6];
  const float* ln_b  = (const float*)d_in[7];
  const float* wqkv  = (const float*)d_in[8];
  const float* wo    = (const float*)d_in[9];
  const float* bo    = (const float*)d_in[10];
  const float* up_w0 = (const float*)d_in[11];
  const float* up_b0 = (const float*)d_in[12];
  const float* cw0a  = (const float*)d_in[13];
  const float* g0a   = (const float*)d_in[14];
  const float* b0a   = (const float*)d_in[15];
  const float* cw0b  = (const float*)d_in[16];
  const float* g0b   = (const float*)d_in[17];
  const float* b0b   = (const float*)d_in[18];
  const float* up_w1 = (const float*)d_in[19];
  const float* up_b1 = (const float*)d_in[20];
  const float* cw1a  = (const float*)d_in[21];
  const float* g1a   = (const float*)d_in[22];
  const float* b1a   = (const float*)d_in[23];
  const float* cw1b  = (const float*)d_in[24];
  const float* g1b   = (const float*)d_in[25];
  const float* b1b   = (const float*)d_in[26];
  const float* up_w2 = (const float*)d_in[27];
  const float* up_b2 = (const float*)d_in[28];
  const float* cw2a  = (const float*)d_in[29];
  const float* g2a   = (const float*)d_in[30];
  const float* b2a   = (const float*)d_in[31];
  const float* cw2b  = (const float*)d_in[32];
  const float* g2b   = (const float*)d_in[33];
  const float* b2b   = (const float*)d_in[34];
  const float* lw    = (const float*)d_in[35];
  const float* lb    = (const float*)d_in[36];
  float* out = (float*)d_out;

  char* w8 = (char*)d_ws;
  const size_t MB = 1024ull * 1024ull;
  float*     tokA   = (float*)(w8 + 0 * MB);        // 8 MB
  float*     tokB   = (float*)(w8 + 8 * MB);        // 8 MB
  _Float16*  xn_h   = (_Float16*)(w8 + 16 * MB);    // 4 MB
  _Float16*  qkv_h  = (_Float16*)(w8 + 20 * MB);    // 12 MB
  _Float16*  obuf_h = (_Float16*)(w8 + 32 * MB);    // 4 MB
  _Float16*  wqkvh  = (_Float16*)(w8 + 36 * MB);    // 19 MB
  _Float16*  woh    = (_Float16*)(w8 + 55 * MB);    // 7 MB
  _Float16*  cb_h   = (_Float16*)(w8 + 62 * MB);    // 2 MB
  float*     cnrm   = (float*)(w8 + 64 * MB);
  int*       idxb   = (int*)(w8 + 64 * MB + 16384);
  float*     dots   = (float*)(w8 + 65 * MB);       // 32 MB
  _Float16*  tok_h  = (_Float16*)(w8 + 97 * MB);    // 4 MB
  _Float16*  y0h    = (_Float16*)(w8 + 101 * MB);   // 4 MB
  _Float16*  cw0ah  = (_Float16*)(w8 + 105 * MB);   // 2.4 MB
  _Float16*  cw0bh  = (_Float16*)(w8 + 108 * MB);   // 1.2 MB
  _Float16*  cw1ah  = (_Float16*)(w8 + 110 * MB);   // 0.6 MB
  _Float16*  cw1bh  = (_Float16*)(w8 + 111 * MB);   // 0.3 MB
  _Float16*  cw2ah  = (_Float16*)(w8 + 112 * MB);   // 0.15 MB
  _Float16*  cw2bh  = (_Float16*)(w8 + 113 * MB);   // 0.08 MB
  _Float16*  u_h    = (_Float16*)(w8 + 114 * MB);   // 67 MB (reused per stage)
  float*     z32    = (float*)(w8 + 181 * MB);      // 67 MB (reused per conv)
  _Float16*  zhA    = (_Float16*)(w8 + 248 * MB);   // 34 MB
  _Float16*  zhB    = (_Float16*)(w8 + 282 * MB);   // 34 MB
  float*     bnm    = (float*)(w8 + 316 * MB);
  float*     bni    = (float*)(w8 + 316 * MB + 4096);

  const int ROWS = NB_ * NTOK_;  // 4096

  // ---- one-time weight conversion / repack (f32 -> f16) ----
  cvt_f16<<<(int)cdl((long)DEPTH_ * D_ * 1536, 256), 256, 0, stream>>>(
      wqkv, wqkvh, (long)DEPTH_ * D_ * 1536);
  cvt_f16<<<(int)cdl((long)DEPTH_ * D_ * D_, 256), 256, 0, stream>>>(
      wo, woh, (long)DEPTH_ * D_ * D_);
  cvt_f16<<<(int)cdl((long)KCODE_ * D_, 256), 256, 0, stream>>>(cb, cb_h,
                                                                (long)KCODE_ * D_);
  repack_w9<<<(int)cdl((long)256 * 512 * 9, 256), 256, 0, stream>>>(cw0a, cw0ah, 256, 512);
  repack_w9<<<(int)cdl((long)256 * 256 * 9, 256), 256, 0, stream>>>(cw0b, cw0bh, 256, 256);
  repack_w9<<<(int)cdl((long)128 * 256 * 9, 256), 256, 0, stream>>>(cw1a, cw1ah, 128, 256);
  repack_w9<<<(int)cdl((long)128 * 128 * 9, 256), 256, 0, stream>>>(cw1b, cw1bh, 128, 128);
  repack_w9<<<(int)cdl((long)64 * 128 * 9, 256), 256, 0, stream>>>(cw2a, cw2ah, 64, 128);
  repack_w9<<<(int)cdl((long)64 * 64 * 9, 256), 256, 0, stream>>>(cw2b, cw2bh, 64, 64);

  // ---- encoder ----
  tok_init<<<(int)cdl((long)ROWS * D_, 256), 256, 0, stream>>>(x, pos, tokA);
  for (int l = 0; l < DEPTH_; ++l) {
    float* src = (l & 1) ? tokB : tokA;
    float* dst = (l & 1) ? tokA : tokB;
    ln_kernel<<<ROWS, 256, 0, stream>>>(src, ln_g + (size_t)l * D_,
                                        ln_b + (size_t)l * D_, xn_h);
    gemm_h<false, false, false, true><<<dim3(ROWS / 64, 1536 / 128), 128, 0, stream>>>(
        xn_h, D_, wqkvh + (size_t)l * D_ * 1536, 1536, nullptr, nullptr, nullptr,
        qkv_h, 1536, ROWS, 1536, D_);
    attn_wmma<<<NB_ * HEADS_, 512, 0, stream>>>(qkv_h, obuf_h);
    gemm_h<true, true, false, false><<<dim3(ROWS / 64, D_ / 128), 128, 0, stream>>>(
        obuf_h, D_, woh + (size_t)l * D_ * D_, D_, bo + (size_t)l * D_, src, dst,
        nullptr, D_, ROWS, D_, D_);
  }
  float* tok = tokA;  // DEPTH_ even -> result back in tokA

  // ---- VQ: argmin_c (||c||^2 - 2 f.c) then gather ----
  code_norm<<<KCODE_, 128, 0, stream>>>(cb, cnrm);
  cvt_f16<<<(int)cdl((long)ROWS * D_, 256), 256, 0, stream>>>(tok, tok_h,
                                                              (long)ROWS * D_);
  gemm_h<false, false, true, false><<<dim3(ROWS / 64, KCODE_ / 128), 128, 0, stream>>>(
      tok_h, D_, cb_h, D_, nullptr, nullptr, dots, nullptr, KCODE_, ROWS, KCODE_, D_);
  argmin_kernel<<<ROWS, 256, 0, stream>>>(dots, cnrm, idxb);
  gather_q<<<(int)cdl((long)ROWS * D_, 256), 256, 0, stream>>>(cb_h, idxb, y0h);

  // ---- decoder ----
  // stage 0: 16x16 -> 32x32, concat feat2 (256+256=512 ch)
  convt2_h<<<(int)cdl((long)NB_ * 256 * 1024, 256), 256, 0, stream>>>(
      y0h, up_w0, up_b0, u_h, NB_, 512, 256, 16, 16, 512);
  copy_skip_h<<<(int)cdl((long)NB_ * 256 * 1024, 256), 256, 0, stream>>>(
      feat2, u_h, NB_, 256, 256, 512, 1024);
  conv3_h<<<dim3(4, (int)cdl((long)NB_ * 1024, 128)), 128, 0, stream>>>(
      u_h, cw0ah, z32, NB_, 512, 256, 32, 32);
  bn_stats<<<256, 256, 0, stream>>>(z32, bnm, bni, NB_, 256, 1024);
  bn_apply_relu_h<<<(int)cdl((long)NB_ * 256 * 1024, 256), 256, 0, stream>>>(
      z32, bnm, bni, g0a, b0a, zhA, 256, 1024, (long)NB_ * 256 * 1024);
  conv3_h<<<dim3(4, (int)cdl((long)NB_ * 1024, 128)), 128, 0, stream>>>(
      zhA, cw0bh, z32, NB_, 256, 256, 32, 32);
  bn_stats<<<256, 256, 0, stream>>>(z32, bnm, bni, NB_, 256, 1024);
  bn_apply_relu_h<<<(int)cdl((long)NB_ * 256 * 1024, 256), 256, 0, stream>>>(
      z32, bnm, bni, g0b, b0b, zhB, 256, 1024, (long)NB_ * 256 * 1024);

  // stage 1: 32x32 -> 64x64, concat feat1 (128+128=256 ch)
  convt2_h<<<(int)cdl((long)NB_ * 128 * 4096, 256), 256, 0, stream>>>(
      zhB, up_w1, up_b1, u_h, NB_, 256, 128, 32, 32, 256);
  copy_skip_h<<<(int)cdl((long)NB_ * 128 * 4096, 256), 256, 0, stream>>>(
      feat1, u_h, NB_, 128, 128, 256, 4096);
  conv3_h<<<dim3(2, (int)cdl((long)NB_ * 4096, 128)), 128, 0, stream>>>(
      u_h, cw1ah, z32, NB_, 256, 128, 64, 64);
  bn_stats<<<128, 256, 0, stream>>>(z32, bnm, bni, NB_, 128, 4096);
  bn_apply_relu_h<<<(int)cdl((long)NB_ * 128 * 4096, 256), 256, 0, stream>>>(
      z32, bnm, bni, g1a, b1a, zhA, 128, 4096, (long)NB_ * 128 * 4096);
  conv3_h<<<dim3(2, (int)cdl((long)NB_ * 4096, 128)), 128, 0, stream>>>(
      zhA, cw1bh, z32, NB_, 128, 128, 64, 64);
  bn_stats<<<128, 256, 0, stream>>>(z32, bnm, bni, NB_, 128, 4096);
  bn_apply_relu_h<<<(int)cdl((long)NB_ * 128 * 4096, 256), 256, 0, stream>>>(
      z32, bnm, bni, g1b, b1b, zhB, 128, 4096, (long)NB_ * 128 * 4096);

  // stage 2: 64x64 -> 128x128, concat feat0 (64+64=128 ch)
  convt2_h<<<(int)cdl((long)NB_ * 64 * 16384, 256), 256, 0, stream>>>(
      zhB, up_w2, up_b2, u_h, NB_, 128, 64, 64, 64, 128);
  copy_skip_h<<<(int)cdl((long)NB_ * 64 * 16384, 256), 256, 0, stream>>>(
      feat0, u_h, NB_, 64, 64, 128, 16384);
  conv3_h<<<dim3(1, (int)cdl((long)NB_ * 16384, 128)), 128, 0, stream>>>(
      u_h, cw2ah, z32, NB_, 128, 64, 128, 128);
  bn_stats<<<64, 256, 0, stream>>>(z32, bnm, bni, NB_, 64, 16384);
  bn_apply_relu_h<<<(int)cdl((long)NB_ * 64 * 16384, 256), 256, 0, stream>>>(
      z32, bnm, bni, g2a, b2a, zhA, 64, 16384, (long)NB_ * 64 * 16384);
  conv3_h<<<dim3(1, (int)cdl((long)NB_ * 16384, 128)), 128, 0, stream>>>(
      zhA, cw2bh, z32, NB_, 64, 64, 128, 128);
  bn_stats<<<64, 256, 0, stream>>>(z32, bnm, bni, NB_, 64, 16384);
  bn_apply_relu_h<<<(int)cdl((long)NB_ * 64 * 16384, 256), 256, 0, stream>>>(
      z32, bnm, bni, g2b, b2b, zhB, 64, 16384, (long)NB_ * 64 * 16384);

  // final 1x1 conv + sigmoid
  final_conv_sigmoid<<<(int)cdl((long)NB_ * 16384, 256), 256, 0, stream>>>(
      zhB, lw, lb, out, NB_, 64, 16384);
}
